// WindowAttention_32530082300576
// MI455X (gfx1250) — compile-verified, matrix operand order checked by reading, not run on hardware
//
#include <hip/hip_runtime.h>
#include <hip/hip_bf16.h>

typedef _Float16 half8 __attribute__((ext_vector_type(8)));
typedef _Float16 v16h  __attribute__((ext_vector_type(16)));
typedef float    v8f   __attribute__((ext_vector_type(8)));

#define DIMC   192
#define NTOK   49
#define HEADS  6
#define HD     32
#define SCALE  0.17677669529663689f   // 32^-0.5
#define XST    208                    // halves per row for 192-wide LDS arrays (16B-aligned, bank-spread)
#define VTST   72                     // halves per row for 64-wide LDS arrays
#define AST    72

// ---- LDS byte offsets (dynamic shared) ----
#define OFF_XS    0                       // x f16 [64][XST]        (reused as outh after stage 2)
#define OFF_QS    26624                   // q f16 [64][XST] (pre-scaled)
#define OFF_KS    53248                   // k f16 [64][XST]
#define OFF_VT    79872                   // v^T f16 [192][VTST]  (dim-major, token inner)
#define OFF_ATTN  107520                  // attn f16 [6][64][AST]
#define SMEM_BYTES 162816

// Load a 16x32 f16 WMMA operand from a row-major LDS buffer.
// A-operand: major = M rows.  B-operand: major = N cols of a K-major buffer.
// A lane's halves 0..7 hold K = kb..kb+7 and halves 8..15 hold K = kb+16..kb+23
// (kb = 8*(lane>=16)) -> two ds_load_b128 per operand.
__device__ __forceinline__ v16h ld_tile(const _Float16* buf, int major0, int stride, int k0) {
    int lane = threadIdx.x & 31;
    const _Float16* p = buf + (major0 + (lane & 15)) * stride + k0 + ((lane >> 4) << 3);
    union { v16h v; half8 h[2]; } u;
    u.h[0] = *(const half8*)p;
    u.h[1] = *(const half8*)(p + 16);
    return u.v;
}

// Load a pre-swizzled 32x16 weight B-tile: tile*512 halves, part-major then lane.
__device__ __forceinline__ v16h ld_w(const _Float16* wsw, int tile) {
    int lane = threadIdx.x & 31;
    const _Float16* p = wsw + tile * 512 + lane * 8;
    union { v16h v; half8 h[2]; } u;
    u.h[0] = *(const half8*)p;
    u.h[1] = *(const half8*)(p + 256);
    return u.v;
}

__device__ __forceinline__ v8f wmma_f32(v16h a, v16h b, v8f c) {
    return __builtin_amdgcn_wmma_f32_16x16x32_f16(false, a, false, b, (short)0, c, false, false);
}

// ---------------- prep kernels ----------------
// Swizzle fp32 weights [K][Ncols] row-major -> f16 B-operand order.
// dst halves layout: tile(=nt*KT+kt)*512 + part*256 + lane*8 + e
// element K = kt*32 + part*16 + 8*(lane>=16) + e ; N = nt*16 + (lane&15)
__global__ void prep_swz(const float* __restrict__ w, _Float16* __restrict__ dst,
                         int ncols, int kt_tiles, int total) {
    int idx = blockIdx.x * 256 + threadIdx.x;
    if (idx >= total) return;
    int tile = idx >> 9, r = idx & 511;
    int nt = tile / kt_tiles, kt = tile % kt_tiles;
    int part = r >> 8, lane = (r >> 3) & 31, e = r & 7;
    int K = kt * 32 + part * 16 + ((lane >> 4) << 3) + e;
    int N = nt * 16 + (lane & 15);
    dst[idx] = (_Float16)w[K * ncols + N];
}

// biasmask swizzled to C-accumulator order:
// dst[(((wg*6 + h)*16 + tile)*32 + lane)*8 + r] where tile = mt*4+nt,
// i = mt*16 + 8*(lane>=16) + r , j = nt*16 + (lane&15).  -inf for j>=49.
__global__ void prep_biasmask(const float* __restrict__ table, const float* __restrict__ mask,
                              float* __restrict__ dst) {
    int idx = blockIdx.x * 256 + threadIdx.x;   // 64*6*16*32*8 total
    if (idx >= 64 * HEADS * 4096) return;
    int r    = idx & 7;
    int lane = (idx >> 3) & 31;
    int t    = (idx >> 8) & 15;
    int h    = (idx >> 12) % HEADS;
    int wg   = idx / (HEADS * 4096);
    int i = (t >> 2) * 16 + ((lane >> 4) << 3) + r;
    int j = (t & 3) * 16 + (lane & 15);
    float v;
    if (j >= NTOK)      v = -1e30f;
    else if (i >= NTOK) v = 0.f;
    else {
        int rel = ((i % 7) - (j % 7) + 6) * 13 + ((i / 7) - (j / 7) + 6);
        v = table[rel * HEADS + h] + mask[(wg * NTOK + i) * NTOK + j];
    }
    dst[idx] = v;
}

// ---------------- fused window-attention kernel ----------------
__global__ __launch_bounds__(192) void win_attn(
    const float* __restrict__ x, const float* __restrict__ b_qkv,
    const float* __restrict__ b_proj, const _Float16* __restrict__ wqkv,
    const _Float16* __restrict__ wproj, const float* __restrict__ biasmask,
    float* __restrict__ out) {
    extern __shared__ char smem[];
    _Float16* xs   = (_Float16*)(smem + OFF_XS);
    _Float16* qs   = (_Float16*)(smem + OFF_QS);
    _Float16* ks   = (_Float16*)(smem + OFF_KS);
    _Float16* vt   = (_Float16*)(smem + OFF_VT);
    _Float16* attn = (_Float16*)(smem + OFF_ATTN);
    _Float16* outh = xs;   // reuse xs after stage 2

    const int b      = blockIdx.x;
    const int tid    = threadIdx.x;
    const int wv     = __builtin_amdgcn_readfirstlane(tid >> 5);  // scalar wave id 0..5
    const int lane   = tid & 31;
    const int lanen  = lane & 15;
    const int halfid = lane >> 4;

    // ---- stage 1: x -> LDS f16, zero-pad rows 49..63 ----
    const float* xb = x + (size_t)b * NTOK * DIMC;
    for (int row = 0; row < 64; ++row)
        xs[row * XST + tid] = (row < NTOK) ? (_Float16)xb[row * DIMC + tid] : (_Float16)0.f;
    __syncthreads();

    // ---- stage 2: qkv = x @ Wqkv + b ; q(*scale), k row-major, v transposed ----
    #pragma unroll 1
    for (int mt = 0; mt < 4; ++mt) {
        v16h A[6];
        #pragma unroll
        for (int kt = 0; kt < 6; ++kt) A[kt] = ld_tile(xs, mt * 16, XST, kt * 32);
        #pragma unroll 1
        for (int j = 0; j < 6; ++j) {
            int nt = wv * 6 + j;                 // scalar: 36 N-tiles over 6 waves
            v8f acc = {};
            #pragma unroll
            for (int kt = 0; kt < 6; ++kt)
                acc = wmma_f32(A[kt], ld_w(wqkv, nt * 6 + kt), acc);
            float bias = b_qkv[nt * 16 + lanen];
            int row0 = mt * 16 + halfid * 8;
            if (nt < 12) {                       // scalar branch: q (pre-scaled)
                int cl = nt * 16 + lanen;
                #pragma unroll
                for (int r = 0; r < 8; ++r)
                    qs[(row0 + r) * XST + cl] = (_Float16)((acc[r] + bias) * SCALE);
            } else if (nt < 24) {                // k
                int cl = nt * 16 - DIMC + lanen;
                #pragma unroll
                for (int r = 0; r < 8; ++r)
                    ks[(row0 + r) * XST + cl] = (_Float16)(acc[r] + bias);
            } else {                             // v, stored transposed (dim-major)
                int cl = nt * 16 - 2 * DIMC + lanen;
                #pragma unroll
                for (int r = 0; r < 8; ++r)
                    vt[cl * VTST + row0 + r] = (_Float16)(acc[r] + bias);
            }
        }
    }
    __syncthreads();

    // ---- stage 3: scores = qk^T + bias+mask ; softmax ; attn -> LDS f16 (wave = head) ----
    const int h = wv;
    const float* bm = biasmask + ((size_t)((b & 63) * HEADS + h)) * 4096 + lane * 8;
    _Float16* ah = attn + h * 64 * AST;
    #pragma unroll 1
    for (int mt = 0; mt < 4; ++mt) {
        v16h A = ld_tile(qs, mt * 16, XST, h * HD);
        v8f s[4];
        #pragma unroll
        for (int nt = 0; nt < 4; ++nt) {
            v8f c = *(const v8f*)(bm + (mt * 4 + nt) * 256);   // 32B vector init of C
            s[nt] = wmma_f32(A, ld_tile(ks, nt * 16, XST, h * HD), c);
        }
        #pragma unroll
        for (int r = 0; r < 8; ++r) {
            float m = fmaxf(fmaxf(s[0][r], s[1][r]), fmaxf(s[2][r], s[3][r]));
            m = fmaxf(m, __shfl_xor(m, 8, 32));
            m = fmaxf(m, __shfl_xor(m, 4, 32));
            m = fmaxf(m, __shfl_xor(m, 2, 32));
            m = fmaxf(m, __shfl_xor(m, 1, 32));
            float e0 = __expf(s[0][r] - m), e1 = __expf(s[1][r] - m);
            float e2 = __expf(s[2][r] - m), e3 = __expf(s[3][r] - m);
            float sum = (e0 + e1) + (e2 + e3);
            sum += __shfl_xor(sum, 8, 32);
            sum += __shfl_xor(sum, 4, 32);
            sum += __shfl_xor(sum, 2, 32);
            sum += __shfl_xor(sum, 1, 32);
            float inv = __builtin_amdgcn_rcpf(sum);
            int row = mt * 16 + halfid * 8 + r;
            ah[row * AST +  0 + lanen] = (_Float16)(e0 * inv);
            ah[row * AST + 16 + lanen] = (_Float16)(e1 * inv);
            ah[row * AST + 32 + lanen] = (_Float16)(e2 * inv);
            ah[row * AST + 48 + lanen] = (_Float16)(e3 * inv);
        }
    }
    // no barrier needed: attn is wave-local, vt covered by stage-2 barrier

    // ---- stage 4: out_h = attn @ v  (B operand = vt, dim-major) ----
    #pragma unroll 1
    for (int mt = 0; mt < 4; ++mt) {
        v16h A0 = ld_tile(ah, mt * 16, AST, 0);
        v16h A1 = ld_tile(ah, mt * 16, AST, 32);
        #pragma unroll
        for (int nt2 = 0; nt2 < 2; ++nt2) {
            v8f acc = {};
            acc = wmma_f32(A0, ld_tile(vt, h * HD + nt2 * 16, VTST, 0),  acc);
            acc = wmma_f32(A1, ld_tile(vt, h * HD + nt2 * 16, VTST, 32), acc);
            #pragma unroll
            for (int r = 0; r < 8; ++r) {
                int row = mt * 16 + halfid * 8 + r;
                outh[row * XST + h * HD + nt2 * 16 + lanen] = (_Float16)acc[r];
            }
        }
    }
    __syncthreads();

    // ---- stage 5: out = outh @ Wproj + b_proj (fp32 store, rows < 49) ----
    float* ob = out + (size_t)b * NTOK * DIMC;
    #pragma unroll 1
    for (int mt = 0; mt < 4; ++mt) {
        v16h A[6];
        #pragma unroll
        for (int kt = 0; kt < 6; ++kt) A[kt] = ld_tile(outh, mt * 16, XST, kt * 32);
        #pragma unroll 1
        for (int jj = 0; jj < 2; ++jj) {
            int nt = wv * 2 + jj;                // scalar: 12 N-tiles over 6 waves
            v8f acc = {};
            #pragma unroll
            for (int kt = 0; kt < 6; ++kt)
                acc = wmma_f32(A[kt], ld_w(wproj, nt * 6 + kt), acc);
            int col = nt * 16 + lanen;
            float bias = b_proj[col];
            #pragma unroll
            for (int r = 0; r < 8; ++r) {
                int row = mt * 16 + halfid * 8 + r;
                if (row < NTOK) ob[row * DIMC + col] = acc[r] + bias;
            }
        }
    }
}

extern "C" void kernel_launch(void* const* d_in, const int* in_sizes, int n_in,
                              void* d_out, int out_size, void* d_ws, size_t ws_size,
                              hipStream_t stream) {
    const float* x      = (const float*)d_in[0];
    const float* mask   = (const float*)d_in[1];
    const float* w_qkv  = (const float*)d_in[2];
    const float* b_qkv  = (const float*)d_in[3];
    const float* w_proj = (const float*)d_in[4];
    const float* b_proj = (const float*)d_in[5];
    const float* table  = (const float*)d_in[6];
    float* out = (float*)d_out;

    _Float16* wqkv_sw  = (_Float16*)d_ws;                       // 36*6*512 halves = 221184 B
    _Float16* wproj_sw = (_Float16*)((char*)d_ws + 221184);     // 12*6*512 halves =  73728 B
    float*    biasmask = (float*)((char*)d_ws + 294912);        // 64*6*4096 f32   = 6291456 B

    prep_swz<<<(110592 + 255) / 256, 256, 0, stream>>>(w_qkv, wqkv_sw, 576, 6, 110592);
    prep_swz<<<(36864  + 255) / 256, 256, 0, stream>>>(w_proj, wproj_sw, 192, 6, 36864);
    prep_biasmask<<<(1572864 + 255) / 256, 256, 0, stream>>>(table, mask, biasmask);
    win_attn<<<4096, 192, SMEM_BYTES, stream>>>(x, b_qkv, b_proj, wqkv_sw, wproj_sw, biasmask, out);
}
